// ManifoldConstrainedHyperConnection_50105088475587
// MI455X (gfx1250) — matrix operand order repaired and checked by reading, not compiled, not run on hardware
//
#include <hip/hip_runtime.h>
#include <hip/hip_bf16.h>

typedef __attribute__((ext_vector_type(16))) __bf16 v16bf;
typedef __attribute__((ext_vector_type(8)))  __bf16 v8bf;
typedef __attribute__((ext_vector_type(4)))  __bf16 v4bf;
typedef __attribute__((ext_vector_type(8)))  float  v8f;

#define NTOK 8192   // B*S
#define NN   4
#define DD   1024

// ---------------------------------------------------------------------------
// K0: transpose W_sub (D,D) f32 -> Wt (D,D) bf16 with Wt[j][k] = W[k][j]
// ---------------------------------------------------------------------------
__global__ __launch_bounds__(256) void k_transpose_w(const float* __restrict__ W,
                                                     __bf16* __restrict__ Wt) {
  int idx = blockIdx.x * 256 + threadIdx.x;      // j*1024 + k
  int j = idx >> 10, k = idx & 1023;
  Wt[idx] = (__bf16)W[k * DD + j];
}

// ---------------------------------------------------------------------------
// K1: LayerNorm (4 streams) + stream-mean + tanh projections ->
//     alphaE = exp(dyn_alpha + static_alpha)  (NTOK x 20), beta (NTOK x 4)
// one token per 256-thread block
// ---------------------------------------------------------------------------
__global__ __launch_bounds__(256) void k_ln_alpha_beta(
    const float* __restrict__ h, const float* __restrict__ SA,
    const float* __restrict__ SB, const float* __restrict__ Wa,
    const float* __restrict__ aScale, const float* __restrict__ Wb,
    const float* __restrict__ bScale, const float* __restrict__ gamma,
    const float* __restrict__ lnbias, float* __restrict__ alphaE,
    float* __restrict__ betaOut) {
  __shared__ float s1[256], s2[256], bc2[2];
  __shared__ float sp[9 * 256];
  const int tid = threadIdx.x;
  const size_t i = blockIdx.x;
  const int d0 = tid * 4;
  const float4 g4 = *(const float4*)(gamma + d0);
  const float4 lb4 = *(const float4*)(lnbias + d0);
  float nm[4] = {0.f, 0.f, 0.f, 0.f};
#pragma unroll
  for (int n = 0; n < NN; ++n) {
    float4 x = *(const float4*)(h + (i * NN + n) * DD + d0);
    s1[tid] = x.x + x.y + x.z + x.w;
    s2[tid] = x.x * x.x + x.y * x.y + x.z * x.z + x.w * x.w;
    __syncthreads();
    for (int st = 128; st > 0; st >>= 1) {
      if (tid < st) { s1[tid] += s1[tid + st]; s2[tid] += s2[tid + st]; }
      __syncthreads();
    }
    if (tid == 0) {
      float mu = s1[0] * (1.f / DD);
      float var = s2[0] * (1.f / DD) - mu * mu;
      bc2[0] = mu; bc2[1] = rsqrtf(var + 1e-5f);
    }
    __syncthreads();
    const float mu = bc2[0], rs = bc2[1];
    nm[0] += 0.25f * ((x.x - mu) * rs * g4.x + lb4.x);
    nm[1] += 0.25f * ((x.y - mu) * rs * g4.y + lb4.y);
    nm[2] += 0.25f * ((x.z - mu) * rs * g4.z + lb4.z);
    nm[3] += 0.25f * ((x.w - mu) * rs * g4.w + lb4.w);
  }
  float pa[5] = {0, 0, 0, 0, 0}, pb[4] = {0, 0, 0, 0};
#pragma unroll
  for (int k = 0; k < 4; ++k) {
    const int d = d0 + k;
    const float v = nm[k];
#pragma unroll
    for (int j = 0; j < 5; ++j) pa[j] += v * Wa[d * 5 + j];
#pragma unroll
    for (int j = 0; j < 4; ++j) pb[j] += v * Wb[d * 4 + j];
  }
#pragma unroll
  for (int j = 0; j < 5; ++j) sp[j * 256 + tid] = pa[j];
#pragma unroll
  for (int j = 0; j < 4; ++j) sp[(5 + j) * 256 + tid] = pb[j];
  __syncthreads();
  for (int st = 128; st > 0; st >>= 1) {
    if (tid < st) {
#pragma unroll
      for (int j = 0; j < 9; ++j) sp[j * 256 + tid] += sp[j * 256 + tid + st];
    }
    __syncthreads();
  }
  if (tid < 20) {
    int n = tid / 5, m = tid % 5;
    float dyn = tanhf(sp[m * 256]) * aScale[0];
    alphaE[i * 20 + tid] = expf(dyn + SA[n * 5 + m]);
  } else if (tid < 24) {
    int n = tid - 20;
    betaOut[i * 4 + n] = tanhf(sp[(5 + n) * 256]) * bScale[0] + SB[n];
  }
}

// ---------------------------------------------------------------------------
// K2: Sinkhorn, single persistent block of 1024 threads (8 rows/thread).
// State m = E * U[n,m] * V[i,m]; col-step updates U (global reduce), row-step
// updates V (row-local). E stays resident in L2 and is re-streamed.
// ---------------------------------------------------------------------------
__global__ __launch_bounds__(1024) void k_sinkhorn(const float* __restrict__ E,
                                                   float* __restrict__ out) {
  __shared__ float red2[32 * 20];
  __shared__ float ush[20];
  const int tid = threadIdx.x;
  const int lane = tid & 31, wave = tid >> 5;
  float V[8][5];
#pragma unroll
  for (int r = 0; r < 8; ++r)
#pragma unroll
    for (int m = 0; m < 5; ++m) V[r][m] = 1.f;
  float u[20];
#pragma unroll
  for (int j = 0; j < 20; ++j) u[j] = 1.f;

  for (int it = 0; it < 20; ++it) {
    float t[20];
#pragma unroll
    for (int j = 0; j < 20; ++j) t[j] = 0.f;
#pragma unroll
    for (int r = 0; r < 8; ++r) {
      const float* e = E + (size_t)(tid * 8 + r) * 20;
      float ev[20];
#pragma unroll
      for (int q = 0; q < 5; ++q) *(float4*)&ev[q * 4] = *(const float4*)(e + q * 4);
#pragma unroll
      for (int n = 0; n < 4; ++n)
#pragma unroll
        for (int m = 0; m < 5; ++m) t[n * 5 + m] += ev[n * 5 + m] * V[r][m];
    }
    // wave32 shuffle reduce, then cross-wave via LDS
#pragma unroll
    for (int j = 0; j < 20; ++j)
      for (int s = 16; s > 0; s >>= 1) t[j] += __shfl_down(t[j], s, 32);
    if (lane == 0) {
#pragma unroll
      for (int j = 0; j < 20; ++j) red2[wave * 20 + j] = t[j];
    }
    __syncthreads();
    if (tid < 20) {
      float s = 0.f;
      for (int w = 0; w < 32; ++w) s += red2[w * 20 + tid];
      ush[tid] = 1.f / s;
    }
    __syncthreads();
#pragma unroll
    for (int j = 0; j < 20; ++j) u[j] = ush[j];
#pragma unroll
    for (int r = 0; r < 8; ++r) {
      const float* e = E + (size_t)(tid * 8 + r) * 20;
      float ev[20];
#pragma unroll
      for (int q = 0; q < 5; ++q) *(float4*)&ev[q * 4] = *(const float4*)(e + q * 4);
#pragma unroll
      for (int m = 0; m < 5; ++m) {
        float sm = ev[m] * u[m] + ev[5 + m] * u[5 + m] + ev[10 + m] * u[10 + m] +
                   ev[15 + m] * u[15 + m];
        V[r][m] = 1.f / sm;
      }
    }
    __syncthreads();
  }
#pragma unroll
  for (int r = 0; r < 8; ++r) {
    const float* e = E + (size_t)(tid * 8 + r) * 20;
    float* o = out + (size_t)(tid * 8 + r) * 20;
    float ev[20];
#pragma unroll
    for (int q = 0; q < 5; ++q) *(float4*)&ev[q * 4] = *(const float4*)(e + q * 4);
#pragma unroll
    for (int n = 0; n < 4; ++n)
#pragma unroll
      for (int m = 0; m < 5; ++m) ev[n * 5 + m] *= u[n * 5 + m] * V[r][m];
#pragma unroll
    for (int q = 0; q < 5; ++q) *(float4*)(o + q * 4) = *(const float4*)&ev[q * 4];
  }
}

// ---------------------------------------------------------------------------
// K3: mix0[i,:] = sum_m alphaMC[i,0,m+1] * h[i,m,:]  -> bf16 (GEMM A matrix)
// ---------------------------------------------------------------------------
__global__ __launch_bounds__(256) void k_mix0(const float* __restrict__ h,
                                              const float* __restrict__ aMC,
                                              __bf16* __restrict__ mix0) {
  const size_t i = blockIdx.x;
  const int d0 = threadIdx.x * 4;
  const float* am = aMC + i * 20;
  const float a1 = am[1], a2 = am[2], a3 = am[3], a4 = am[4];
  const float* hp = h + i * NN * DD + d0;
  float4 x0 = *(const float4*)(hp);
  float4 x1 = *(const float4*)(hp + DD);
  float4 x2 = *(const float4*)(hp + 2 * DD);
  float4 x3 = *(const float4*)(hp + 3 * DD);
  float4 acc;
  acc.x = a1 * x0.x + a2 * x1.x + a3 * x2.x + a4 * x3.x;
  acc.y = a1 * x0.y + a2 * x1.y + a3 * x2.y + a4 * x3.y;
  acc.z = a1 * x0.z + a2 * x1.z + a3 * x2.z + a4 * x3.z;
  acc.w = a1 * x0.w + a2 * x1.w + a3 * x2.w + a4 * x3.w;
  v4bf r;
  r.x = (__bf16)acc.x; r.y = (__bf16)acc.y; r.z = (__bf16)acc.z; r.w = (__bf16)acc.w;
  *(v4bf*)(mix0 + i * DD + d0) = r;
}

// ---------------------------------------------------------------------------
// K4: WMMA GEMM h_o = mix0(bf16) @ Wt^T with fused epilogue
//     out[i,n,j] = beta[i,n]*h_o[i,j] + sum_m alphaMC[i,n,m+1]*h[i,m,j]
// 128x128 block tile, 8 waves of 64x32, K-step 32, double-buffered async LDS.
// ---------------------------------------------------------------------------
__device__ __forceinline__ void async_b128(unsigned ldsOff, const void* gptr) {
  asm volatile("global_load_async_to_lds_b128 %0, %1, off"
               :: "v"(ldsOff), "v"(gptr) : "memory");
}

__global__ __launch_bounds__(256) void k_gemm_out(
    const __bf16* __restrict__ A,      // mix0  [NTOK][DD]
    const __bf16* __restrict__ Bt,     // Wt    [DD(j)][DD(k)]
    const float* __restrict__ h, const float* __restrict__ aMC,
    const float* __restrict__ beta, float* __restrict__ out) {
  __shared__ __bf16 As[2][128 * 32];
  __shared__ __bf16 Bs[2][128 * 32];   // [j][k]
  __shared__ float AB[128 * 24];       // per-token 20 alpha + 4 beta

  const int tid = threadIdx.x;
  const int lane = tid & 31;
  const int w = tid >> 5;
  const int wm = w >> 2;               // 0..1 : 64-row slab
  const int wn = w & 3;                // 0..3 : 32-col slab
  const int m0 = blockIdx.y * 128;
  const int n0 = blockIdx.x * 128;
  const int hl = lane >> 4;            // half-wave select
  const int l15 = lane & 15;

  v8f acc[4][2] = {};

  auto issue = [&](int buf, int k0) {
#pragma unroll
    for (int it = 0; it < 2; ++it) {
      int idx = tid + it * 256;                // 0..511 (chunks of 8 bf16)
      int row = idx >> 2, c8 = (idx & 3) * 8;
      async_b128((unsigned)(size_t)(void*)&As[buf][row * 32 + c8],
                 (const void*)(A + (size_t)(m0 + row) * DD + k0 + c8));
      async_b128((unsigned)(size_t)(void*)&Bs[buf][row * 32 + c8],
                 (const void*)(Bt + (size_t)(n0 + row) * DD + k0 + c8));
    }
  };

  issue(0, 0);
  for (int ks = 0; ks < 32; ++ks) {
    const int buf = ks & 1;
    asm volatile("s_wait_asynccnt 0x0" ::: "memory");
    __syncthreads();
    if (ks + 1 < 32) issue(buf ^ 1, (ks + 1) * 32);

    v16bf bfr[2];
#pragma unroll
    for (int nt = 0; nt < 2; ++nt)
      bfr[nt] = *(const v16bf*)&Bs[buf][(wn * 32 + nt * 16 + l15) * 32 + hl * 16];
#pragma unroll
    for (int mt = 0; mt < 4; ++mt) {
      const int ra = (wm * 64 + mt * 16 + l15) * 32 + hl * 16;
      v8bf alo = *(const v8bf*)&As[buf][ra];
      v8bf ahi = *(const v8bf*)&As[buf][ra + 8];
      v16bf a = __builtin_shufflevector(alo, ahi, 0, 1, 2, 3, 4, 5, 6, 7, 8, 9,
                                        10, 11, 12, 13, 14, 15);
#pragma unroll
      for (int nt = 0; nt < 2; ++nt)
        acc[mt][nt] = __builtin_amdgcn_wmma_f32_16x16x32_bf16(
            false, a, false, bfr[nt], (short)0, acc[mt][nt], false, false);
    }
  }

  // stage alpha/beta for this block's 128 tokens
  for (int q = tid; q < 128 * 24; q += 256) {
    int tok = q / 24, off = q % 24;
    AB[q] = (off < 20) ? aMC[(size_t)(m0 + tok) * 20 + off]
                       : beta[(size_t)(m0 + tok) * 4 + (off - 20)];
  }
  __syncthreads();

#pragma unroll
  for (int mt = 0; mt < 4; ++mt)
#pragma unroll
    for (int nt = 0; nt < 2; ++nt)
#pragma unroll
      for (int r = 0; r < 8; ++r) {
        const int rl = wm * 64 + mt * 16 + r + hl * 8;   // local row (token)
        const int jl = wn * 32 + nt * 16 + l15;          // local col (d)
        const size_t i = m0 + rl;
        const int j = n0 + jl;
        const float ho = acc[mt][nt][r];
        const float* hp = h + i * NN * DD + j;
        const float h0 = hp[0], h1 = hp[DD], h2 = hp[2 * DD], h3 = hp[3 * DD];
        const float* ab = &AB[rl * 24];
#pragma unroll
        for (int n = 0; n < 4; ++n) {
          float mix = ab[n * 5 + 1] * h0 + ab[n * 5 + 2] * h1 +
                      ab[n * 5 + 3] * h2 + ab[n * 5 + 4] * h3;
          out[(i * NN + n) * DD + j] = ab[20 + n] * ho + mix;
        }
      }
}

// ---------------------------------------------------------------------------
extern "C" void kernel_launch(void* const* d_in, const int* in_sizes, int n_in,
                              void* d_out, int out_size, void* d_ws, size_t ws_size,
                              hipStream_t stream) {
  (void)in_sizes; (void)n_in; (void)out_size; (void)ws_size;
  const float* h      = (const float*)d_in[0];
  const float* SA     = (const float*)d_in[1];
  const float* SB     = (const float*)d_in[2];
  const float* Wa     = (const float*)d_in[3];
  const float* aScale = (const float*)d_in[4];
  const float* Wb     = (const float*)d_in[5];
  const float* bScale = (const float*)d_in[6];
  const float* gamma  = (const float*)d_in[7];
  const float* lnbias = (const float*)d_in[8];
  const float* Wsub   = (const float*)d_in[9];
  float* out = (float*)d_out;

  // workspace carve (256-byte aligned)
  char* ws = (char*)d_ws;
  size_t o = 0;
  auto carve = [&](size_t bytes) { void* p = ws + o; o += (bytes + 255) & ~size_t(255); return p; };
  float*  alphaE  = (float*)carve((size_t)NTOK * 20 * 4);
  float*  alphaMC = (float*)carve((size_t)NTOK * 20 * 4);
  float*  betaW   = (float*)carve((size_t)NTOK * 4 * 4);
  __bf16* Wt16    = (__bf16*)carve((size_t)DD * DD * 2);
  __bf16* mix0    = (__bf16*)carve((size_t)NTOK * DD * 2);

  k_transpose_w<<<(DD * DD) / 256, 256, 0, stream>>>(Wsub, Wt16);
  k_ln_alpha_beta<<<NTOK, 256, 0, stream>>>(h, SA, SB, Wa, aScale, Wb, bScale,
                                            gamma, lnbias, alphaE, betaW);
  k_sinkhorn<<<1, 1024, 0, stream>>>(alphaE, alphaMC);
  k_mix0<<<NTOK, 256, 0, stream>>>(h, alphaMC, mix0);
  dim3 gg(DD / 128, NTOK / 128);
  k_gemm_out<<<gg, 256, 0, stream>>>(mix0, Wt16, h, alphaMC, betaW, out);
}
